// WaveFunction_43911745634894
// MI455X (gfx1250) — matrix operand order repaired and checked by reading, not compiled, not run on hardware
//
#include <hip/hip_runtime.h>
#include <hip/hip_bf16.h>
#include <math.h>

// ---------------------------------------------------------------------------
// Autoregressive wavefunction (64 sites, 2 nets/site, MLP [2i -> 512 -> 512 -> 4])
// MI455X / gfx1250:
//   - weights fp32 -> bf16 once, transposed N-major for WMMA B-fragments
//   - main kernel: BM=128 batch rows/block, 512 threads (16 waves);
//     each wave owns 2 N-tiles and reuses each B-fragment across all 8
//     row-groups (B read exactly once per block -> minimal L2 traffic)
//   - activations staged in 256 KB LDS between layers
// ---------------------------------------------------------------------------

typedef __attribute__((ext_vector_type(16))) __bf16 v16bf;
typedef __attribute__((ext_vector_type(8)))  __bf16 v8bf;
typedef __attribute__((ext_vector_type(8)))  float  v8f;

#define SITES 64
#define BATCH 4096
#define HID 512
#define BM 128                // batch rows per block
#define NTILES (BATCH / BM)   // 32 batch tiles
#define K0PAD 128             // padded layer-0 K (max 2*63=126 -> 128)
#define NWAVES 16
#define NTHREADS (NWAVES * 32)
#define NRG (BM / 16)         // 8 row groups per block

// workspace byte offsets
#define OFF_ACC_A   0                        // 4096 f32
#define OFF_ACC_P   16384                    // 4096 f32
#define OFF_BIAS    32768                    // 128 nets * 1040 f32
#define OFF_W2T     565248                   // 128 * 16*512 bf16
#define OFF_W0T     2662400                  // 128 * 512*128 bf16
#define OFF_W1T     19439616                 // 128 * 512*512 bf16

__device__ __forceinline__ __bf16 f2bf(float f) {
    unsigned u = __builtin_bit_cast(unsigned, f);
    unsigned r = (u + 0x7FFFu + ((u >> 16) & 1u)) >> 16;
    return __builtin_bit_cast(__bf16, (unsigned short)r);
}

// 16 bf16 fragment from two 16-byte-aligned 8-element chunks
__device__ __forceinline__ v16bf ld16(const __bf16* p0, const __bf16* p1) {
    v8bf a = *(const v8bf*)p0;
    v8bf b = *(const v8bf*)p1;
    v16bf r;
#pragma unroll
    for (int j = 0; j < 8; ++j) { r[j] = a[j]; r[j + 8] = b[j]; }
    return r;
}

__device__ __forceinline__ v8f wmma_bf16(v16bf a, v16bf b, v8f c) {
    return __builtin_amdgcn_wmma_f32_16x16x32_bf16(
        /*neg_a*/false, a, /*neg_b*/false, b,
        /*c_mod*/(short)0, c, /*reuse_a*/false, /*reuse_b*/false);
}

// ---------------------------------------------------------------------------
__global__ void zero_acc_kernel(float* acc) {
    int t = blockIdx.x * blockDim.x + threadIdx.x;
    if (t < 2 * BATCH) acc[t] = 0.0f;
}

// ---------------------------------------------------------------------------
// per-net weight conversion: fp32 row-major -> bf16 N-major (transposed),
// with K/N zero padding; biases packed as f32 [b0:512 | b1:512 | b2pad:16]
// ---------------------------------------------------------------------------
__global__ void convert_kernel(int site,
                               const float* __restrict__ W0, const float* __restrict__ b0,
                               const float* __restrict__ W1, const float* __restrict__ b1,
                               const float* __restrict__ W2, const float* __restrict__ b2,
                               __bf16* __restrict__ W0T, __bf16* __restrict__ W1T,
                               __bf16* __restrict__ W2T, float* __restrict__ biasDst) {
    const int K2 = 2 * site;
    size_t tid = (size_t)blockIdx.x * blockDim.x + threadIdx.x;
    size_t nth = (size_t)gridDim.x * blockDim.x;

    for (size_t t = tid; t < (size_t)HID * HID; t += nth) {
        int n = (int)(t >> 9), k = (int)(t & 511);
        W1T[t] = f2bf(W1[(size_t)k * HID + n]);
    }
    for (size_t t = tid; t < (size_t)HID * K0PAD; t += nth) {
        int n = (int)(t >> 7), k = (int)(t & 127);
        W0T[t] = (k < K2) ? f2bf(W0[(size_t)k * HID + n]) : f2bf(0.0f);
    }
    for (size_t t = tid; t < (size_t)16 * HID; t += nth) {
        int n = (int)(t >> 9), k = (int)(t & 511);
        W2T[t] = (n < 4) ? f2bf(W2[(size_t)k * 4 + n]) : f2bf(0.0f);
    }
    for (size_t t = tid; t < HID; t += nth) {
        biasDst[t] = b0[t];
        biasDst[HID + t] = b1[t];
    }
    if (tid < 16) biasDst[2 * HID + tid] = (tid < 4) ? b2[tid] : 0.0f;
}

// ---------------------------------------------------------------------------
// main fused MLP kernel
// grid: 128 nets * 32 batch tiles; block: 512 threads = 16 waves
// wave w owns N-tiles {2w, 2w+1}; per B-fragment it sweeps all 8 row groups.
// LDS (256 KB static): sH1[128*512] bf16 | sB: ctx(128*128 bf16) / H2(128*512 bf16)
// out4 (128*16 f32) reuses sH1 region during layer 2.
// ---------------------------------------------------------------------------
__global__ __launch_bounds__(NTHREADS) void mlp_site_kernel(
        const int* __restrict__ x,
        const __bf16* __restrict__ wsW0T, const __bf16* __restrict__ wsW1T,
        const __bf16* __restrict__ wsW2T, const float* __restrict__ wsBias,
        float* __restrict__ accA, float* __restrict__ accP) {
    __shared__ __align__(16) __bf16 sH1[BM * HID];           // 128 KB
    __shared__ __align__(16) unsigned char sBmem[BM * HID * 2]; // 128 KB (ctx / H2)

    __bf16* ctx = (__bf16*)sBmem;   // [128][128] during layer 0
    __bf16* sH2 = (__bf16*)sBmem;   // [128][512] during layers 1/2

    const int blk  = blockIdx.x;
    const int tile = blk & (NTILES - 1);
    const int nid  = blk >> 5;           // 0..127 (0..63 amp, 64..127 phase)
    const int site = nid & 63;
    const int isph = nid >> 6;
    const int row0 = tile * BM;

    const __bf16* W0T = wsW0T + (size_t)nid * (HID * K0PAD);
    const __bf16* W1T = wsW1T + (size_t)nid * (HID * HID);
    const __bf16* W2T = wsW2T + (size_t)nid * (16 * HID);
    const float*  bias = wsBias + (size_t)nid * 1040;

    const int wave  = threadIdx.x >> 5;
    const int lane  = threadIdx.x & 31;
    const int ln15  = lane & 15;         // A row within group / B,C column
    const int khalf = lane >> 4;

    const int K2  = 2 * site;
    const int KT0 = (K2 + 31) >> 5;      // layer-0 K tiles

    // ---- stage context (0/1 ints -> bf16, zero padded to 128 cols) ----
    for (int t = threadIdx.x; t < BM * K0PAD; t += NTHREADS) {
        int m = t >> 7, k = t & 127;
        float v = (k < K2) ? (float)x[(size_t)(row0 + m) * (2 * SITES) + k] : 0.0f;
        ctx[t] = f2bf(v);
    }
    __syncthreads();

    // ---- layer 0: H1 = silu(ctx @ W0 + b0), K = 2i ----
#pragma unroll
    for (int t = 0; t < 2; ++t) {
        const int nt = wave * 2 + t;
        v8f acc[NRG];
#pragma unroll
        for (int rg = 0; rg < NRG; ++rg) acc[rg] = {};
        for (int kt = 0; kt < KT0; ++kt) {
            const __bf16* bp = &W0T[(size_t)(nt * 16 + ln15) * K0PAD + kt * 32 + khalf * 16];
            v16bf B = ld16(bp, bp + 8);
#pragma unroll
            for (int rg = 0; rg < NRG; ++rg) {
                const __bf16* ap = &ctx[(rg * 16 + ln15) * K0PAD + kt * 32 + khalf * 16];
                v16bf A = ld16(ap, ap + 8);
                acc[rg] = wmma_bf16(A, B, acc[rg]);
            }
        }
        const float bv = bias[nt * 16 + ln15];
#pragma unroll
        for (int rg = 0; rg < NRG; ++rg) {
#pragma unroll
            for (int r = 0; r < 8; ++r) {
                float h = acc[rg][r] + bv;
                h = h / (1.0f + __expf(-h));     // silu
                sH1[(rg * 16 + r + 8 * khalf) * HID + nt * 16 + ln15] = f2bf(h);
            }
        }
    }
    __syncthreads();

    // ---- layer 1: H2 = silu(H1 @ W1 + b1), K = 512 ----
#pragma unroll
    for (int t = 0; t < 2; ++t) {
        const int nt = wave * 2 + t;
        v8f acc[NRG];
#pragma unroll
        for (int rg = 0; rg < NRG; ++rg) acc[rg] = {};
        for (int kt = 0; kt < 16; ++kt) {
            const __bf16* bp = &W1T[(size_t)(nt * 16 + ln15) * HID + kt * 32 + khalf * 16];
            v16bf B = ld16(bp, bp + 8);
#pragma unroll
            for (int rg = 0; rg < NRG; ++rg) {
                const __bf16* ap = &sH1[(rg * 16 + ln15) * HID + kt * 32 + khalf * 16];
                v16bf A = ld16(ap, ap + 8);
                acc[rg] = wmma_bf16(A, B, acc[rg]);
            }
        }
        const float bv = bias[HID + nt * 16 + ln15];
#pragma unroll
        for (int rg = 0; rg < NRG; ++rg) {
#pragma unroll
            for (int r = 0; r < 8; ++r) {
                float h = acc[rg][r] + bv;
                h = h / (1.0f + __expf(-h));
                sH2[(rg * 16 + r + 8 * khalf) * HID + nt * 16 + ln15] = f2bf(h);
            }
        }
    }
    __syncthreads();

    // ---- layer 2: out4 = H2 @ W2 + b2 (N padded to 16); kt = wave (16 tiles) ----
    float* out4 = (float*)sH1;           // [128][16], H1 region is dead now
    for (int t = threadIdx.x; t < BM * 16; t += NTHREADS) out4[t] = 0.0f;
    __syncthreads();
    {
        const int kt = wave;             // exactly 16 K-tiles over 16 waves
        v8f acc[NRG];
        const float binit = (wave == 0) ? bias[2 * HID + ln15] : 0.0f;
#pragma unroll
        for (int rg = 0; rg < NRG; ++rg) {
#pragma unroll
            for (int r = 0; r < 8; ++r) acc[rg][r] = binit;
        }
        const __bf16* bp = &W2T[(size_t)ln15 * HID + kt * 32 + khalf * 16];
        v16bf B = ld16(bp, bp + 8);
#pragma unroll
        for (int rg = 0; rg < NRG; ++rg) {
            const __bf16* ap = &sH2[(rg * 16 + ln15) * HID + kt * 32 + khalf * 16];
            v16bf A = ld16(ap, ap + 8);
            acc[rg] = wmma_bf16(A, B, acc[rg]);
        }
#pragma unroll
        for (int rg = 0; rg < NRG; ++rg) {
#pragma unroll
            for (int r = 0; r < 8; ++r)
                atomicAdd(&out4[(rg * 16 + r + 8 * khalf) * 16 + ln15], acc[rg][r]);
        }
    }
    __syncthreads();

    // ---- epilogue: mask + logsumexp (amp) or gather (phase) ----
    if (threadIdx.x < BM) {
        const int m = threadIdx.x;
        const int row = row0 + m;
        const int* xr = x + (size_t)row * (2 * SITES);
        int up_e = 0, dn_e = 0;
        for (int k = 0; k < site; ++k) { up_e += xr[2 * k]; dn_e += xr[2 * k + 1]; }
        const int idx = xr[2 * site] * 2 + xr[2 * site + 1];
        float d[4];
#pragma unroll
        for (int j = 0; j < 4; ++j) d[j] = out4[m * 16 + j];

        if (isph) {
            atomicAdd(&accP[row], d[idx]);
        } else {
            const int up_h = site - up_e, dn_h = site - dn_e;
            const bool au0 = up_h < (SITES - 32), au1 = up_e < 32;
            const bool ad0 = dn_h < (SITES - 32), ad1 = dn_e < 32;
            const bool ok[4] = { au0 && ad0, au0 && ad1, au1 && ad0, au1 && ad1 };
            float mx = -INFINITY;
#pragma unroll
            for (int j = 0; j < 4; ++j) if (ok[j]) mx = fmaxf(mx, d[j]);
            float s = 0.0f;
#pragma unroll
            for (int j = 0; j < 4; ++j) if (ok[j]) s += __expf(2.0f * (d[j] - mx));
            const float norm = mx + 0.5f * __logf(s);
            atomicAdd(&accA[row], d[idx] - norm);
        }
    }
}

// ---------------------------------------------------------------------------
__global__ void finalize_kernel(const float* __restrict__ accA,
                                const float* __restrict__ accP,
                                float* __restrict__ out) {
    int r = blockIdx.x * blockDim.x + threadIdx.x;
    if (r < BATCH) {
        float a = expf(accA[r]);
        float p = accP[r];
        out[2 * r + 0] = a * cosf(p);
        out[2 * r + 1] = a * sinf(p);
    }
}

// ---------------------------------------------------------------------------
extern "C" void kernel_launch(void* const* d_in, const int* in_sizes, int n_in,
                              void* d_out, int out_size, void* d_ws, size_t ws_size,
                              hipStream_t stream) {
    (void)in_sizes; (void)n_in; (void)out_size; (void)ws_size;

    const int* x = (const int*)d_in[0];
    char* ws = (char*)d_ws;
    float*  accA     = (float*)(ws + OFF_ACC_A);
    float*  accP     = (float*)(ws + OFF_ACC_P);
    float*  biasBase = (float*)(ws + OFF_BIAS);
    __bf16* W2Tb     = (__bf16*)(ws + OFF_W2T);
    __bf16* W0Tb     = (__bf16*)(ws + OFF_W0T);
    __bf16* W1Tb     = (__bf16*)(ws + OFF_W1T);

    zero_acc_kernel<<<(2 * BATCH + 255) / 256, 256, 0, stream>>>(accA);

    for (int nid = 0; nid < 128; ++nid) {
        const int isph = nid >> 6;
        const int site = nid & 63;
        const int base = 1 + isph * (SITES * 6) + site * 6;
        convert_kernel<<<96, 256, 0, stream>>>(
            site,
            (const float*)d_in[base + 0], (const float*)d_in[base + 1],
            (const float*)d_in[base + 2], (const float*)d_in[base + 3],
            (const float*)d_in[base + 4], (const float*)d_in[base + 5],
            W0Tb + (size_t)nid * (HID * K0PAD),
            W1Tb + (size_t)nid * (HID * HID),
            W2Tb + (size_t)nid * (16 * HID),
            biasBase + (size_t)nid * 1040);
    }

    mlp_site_kernel<<<128 * NTILES, NTHREADS, 0, stream>>>(
        x, W0Tb, W1Tb, W2Tb, biasBase, accA, accP);

    finalize_kernel<<<(BATCH + 255) / 256, 256, 0, stream>>>(accA, accP, (float*)d_out);
}
